// MultiHeadAttention_8993661518317
// MI455X (gfx1250) — matrix operand order repaired and checked by reading, not compiled, not run on hardware
//
#include <hip/hip_runtime.h>
#include <hip/hip_bf16.h>
#include <math.h>

#define E_DIM  1024
#define NHEAD  16
#define HDIM   64
#define NBATCH 4
#define SEQ    2048   // S == T

typedef __bf16 bf16;
typedef __attribute__((ext_vector_type(16))) __bf16 v16bf;
typedef __attribute__((ext_vector_type(8)))  float  v8f;

union FragU { v16bf v; uint4 q[2]; };

// ---------------------------------------------------------------------------
// WMMA wrapper: D = A(16x32 bf16) * B(32x16 bf16) + C(16x16 f32)
// ---------------------------------------------------------------------------
__device__ __forceinline__ v8f wmma_bf16(v16bf a, v16bf b, v8f c) {
  return __builtin_amdgcn_wmma_f32_16x16x32_bf16(
      /*neg_a=*/false, a, /*neg_b=*/false, b,
      /*c_mod=*/(short)0, c, /*reuse_a=*/false, /*reuse_b=*/false);
}

// A-matrix fragment (16x32, 16-bit): lane L<16 -> row L, K {0..7, 16..23};
// lane L>=16 -> row L-16, K {8..15, 24..31}.  Two 16B loads per lane.
__device__ __forceinline__ v16bf frag_a_bf16(const bf16* p, int ld, int lane) {
  const int half = (lane >> 4) & 1;
  const int r    = lane & 15;
  const bf16* row = p + (size_t)r * ld;
  FragU f;
  f.q[0] = *(const uint4*)(row + half * 8);
  f.q[1] = *(const uint4*)(row + 16 + half * 8);
  return f.v;
}

// B-matrix fragment (32x16, 16-bit) from a [n][k]-transposed tile:
// lane L<16 -> col L, K {0..7, 16..23}; lane L>=16 -> col L-16, K {8..15, 24..31}.
__device__ __forceinline__ v16bf frag_b_bf16(const bf16* p, int ld, int lane) {
  const int half = (lane >> 4) & 1;
  const int r    = lane & 15;
  const bf16* row = p + (size_t)r * ld;
  FragU f;
  f.q[0] = *(const uint4*)(row + half * 8);
  f.q[1] = *(const uint4*)(row + 16 + half * 8);
  return f.v;
}

// A fragment straight from fp32 activations (convert to bf16 in registers).
__device__ __forceinline__ v16bf frag_a_f32(const float* p, int ld, int lane) {
  const int half = (lane >> 4) & 1;
  const int r    = lane & 15;
  const float* row = p + (size_t)r * ld;
  float4 a0 = *(const float4*)(row + half * 8);
  float4 a1 = *(const float4*)(row + half * 8 + 4);
  float4 b0 = *(const float4*)(row + 16 + half * 8);
  float4 b1 = *(const float4*)(row + 16 + half * 8 + 4);
  v16bf o;
  o[0]=(bf16)a0.x;  o[1]=(bf16)a0.y;  o[2]=(bf16)a0.z;  o[3]=(bf16)a0.w;
  o[4]=(bf16)a1.x;  o[5]=(bf16)a1.y;  o[6]=(bf16)a1.z;  o[7]=(bf16)a1.w;
  o[8]=(bf16)b0.x;  o[9]=(bf16)b0.y;  o[10]=(bf16)b0.z; o[11]=(bf16)b0.w;
  o[12]=(bf16)b1.x; o[13]=(bf16)b1.y; o[14]=(bf16)b1.z; o[15]=(bf16)b1.w;
  return o;
}

// ---------------------------------------------------------------------------
// Shared 64x64 register-blocked GEMM core: 4x4 tiles of 16x16 per wave.
// Per K=32 step: 8 fragment loads feed 16 independent WMMAs (32 FLOP/byte for bf16 A).
// ---------------------------------------------------------------------------
template <bool A_IS_F32>
__device__ __forceinline__ void gemm_core64(const void* Abase, const bf16* Wb,
                                            int lane, v8f acc[4][4]) {
#pragma unroll
  for (int i = 0; i < 4; ++i)
#pragma unroll
    for (int j = 0; j < 4; ++j)
#pragma unroll
      for (int v = 0; v < 8; ++v) acc[i][j][v] = 0.f;

  for (int k0 = 0; k0 < E_DIM; k0 += 32) {
    v16bf a[4], b[4];
#pragma unroll
    for (int i = 0; i < 4; ++i) {
      if constexpr (A_IS_F32)
        a[i] = frag_a_f32((const float*)Abase + (size_t)(i*16)*E_DIM + k0, E_DIM, lane);
      else
        a[i] = frag_a_bf16((const bf16*)Abase + (size_t)(i*16)*E_DIM + k0, E_DIM, lane);
    }
#pragma unroll
    for (int j = 0; j < 4; ++j)
      b[j] = frag_b_bf16(Wb + (size_t)(j*16)*E_DIM + k0, E_DIM, lane);
#pragma unroll
    for (int i = 0; i < 4; ++i)
#pragma unroll
      for (int j = 0; j < 4; ++j)
        acc[i][j] = wmma_bf16(a[i], b[j], acc[i][j]);
  }
}

// ---------------------------------------------------------------------------
// Weight prep: W[k][c] fp32 -> Wt[c][k] bf16 (B-fragment friendly layout).
// ---------------------------------------------------------------------------
__global__ void prep_weight_kernel(const float* __restrict__ W, bf16* __restrict__ Wt) {
  int idx = blockIdx.x * blockDim.x + threadIdx.x;   // over E*E
  int k = idx >> 10;        // in-dim
  int c = idx & (E_DIM-1);  // out-dim (coalesced reads)
  Wt[(size_t)c * E_DIM + k] = (bf16)W[idx];
}

// ---------------------------------------------------------------------------
// Projection GEMM: 64x64 tile per wave.  C = (X @ W + b) * alpha.
// mode 0: store Q  -> [N,H,S,D] ; mode 1: store K -> [N,H,T,D]
// mode 2: store Vt -> [N,H,D,T]
// ---------------------------------------------------------------------------
__global__ void __launch_bounds__(256, 2)
proj_gemm_kernel(const float* __restrict__ X, const bf16* __restrict__ Wt,
                 const float* __restrict__ bias, bf16* __restrict__ out,
                 int mrows, int mode, float alpha) {
  const int lane = threadIdx.x & 31;
  const int wid  = threadIdx.x >> 5;
  const int gw   = blockIdx.x * 8 + wid;
  const int wm = gw >> 4;            // E/64 = 16 column groups
  const int wn = gw & 15;
  if (wm * 64 >= mrows) return;

  const float* Xt = X  + (size_t)wm * 64 * E_DIM;
  const bf16*  Wb = Wt + (size_t)wn * 64 * E_DIM;

  v8f acc[4][4];
  gemm_core64<true>(Xt, Wb, lane, acc);

  const int half = lane >> 4, nidx = lane & 15;
#pragma unroll
  for (int j = 0; j < 4; ++j) {
    const int col = wn * 64 + j * 16 + nidx;
    const float bc = bias[col];
    const int hh = col >> 6, dd = col & (HDIM-1);
#pragma unroll
    for (int i = 0; i < 4; ++i) {
#pragma unroll
      for (int v = 0; v < 8; ++v) {
        int row = wm * 64 + i * 16 + v + half * 8;
        int nb  = row >> 11;          // / SEQ
        int tok = row & (SEQ-1);
        bf16 bvv = (bf16)((acc[i][j][v] + bc) * alpha);
        if (mode == 2)
          out[(((size_t)nb*NHEAD + hh)*HDIM + dd)*SEQ + tok] = bvv;   // V transposed
        else
          out[(((size_t)nb*NHEAD + hh)*SEQ + tok)*HDIM + dd] = bvv;   // Q / K
      }
    }
  }
}

// ---------------------------------------------------------------------------
// Flash attention: one wave per (n, h, 32-query tile); 32 keys per iteration.
// K/V fragments shared across 2 q-tiles -> 16 WMMAs per 8KB of fragments.
// ---------------------------------------------------------------------------
#define PSTRIDE 40   // padded LDS row stride (bf16 elems), 16B-aligned chunks
__global__ void __launch_bounds__(256, 2)
flash_attn_kernel(const bf16* __restrict__ Qh, const bf16* __restrict__ Kh,
                  const bf16* __restrict__ Vt, bf16* __restrict__ Y) {
  __shared__ bf16 plds[8 * 2 * 16 * PSTRIDE];          // 20 KB / workgroup
  const int lane = threadIdx.x & 31;
  const int wid  = threadIdx.x >> 5;
  const int w    = blockIdx.x * 8 + wid;
  const int n  = w >> 10;        // / (H * S/32)
  const int h  = (w >> 6) & 15;
  const int qt = w & 63;         // 32-query tile
  const int half = lane >> 4, nidx = lane & 15;

  const bf16* Qb = Qh + (((size_t)n*NHEAD + h)*SEQ + qt*32) * HDIM;
  const bf16* Kb = Kh + ((size_t)n*NHEAD + h) * SEQ * HDIM;
  const bf16* Vb = Vt + ((size_t)n*NHEAD + h) * (size_t)HDIM * SEQ;
  bf16* pl = plds + wid * (2 * 16 * PSTRIDE);

  // Q preloaded once (already scaled by 1/sqrt(D) at projection time)
  v16bf qa[2][2];
#pragma unroll
  for (int q = 0; q < 2; ++q) {
    qa[q][0] = frag_a_bf16(Qb + (size_t)q*16*HDIM,      HDIM, lane);
    qa[q][1] = frag_a_bf16(Qb + (size_t)q*16*HDIM + 32, HDIM, lane);
  }

  float mrun[2][8], lrun[2][8];
  v8f acc[2][4];
#pragma unroll
  for (int q = 0; q < 2; ++q)
#pragma unroll
    for (int v = 0; v < 8; ++v) { mrun[q][v] = -1e30f; lrun[q][v] = 0.f; }
#pragma unroll
  for (int q = 0; q < 2; ++q)
#pragma unroll
    for (int t = 0; t < 4; ++t)
#pragma unroll
      for (int v = 0; v < 8; ++v) acc[q][t][v] = 0.f;

  for (int kb = 0; kb < SEQ; kb += 32) {
    if (kb + 32 < SEQ) {   // prefetch next K / V block (global_prefetch_b8)
      __builtin_prefetch(Kb + (size_t)(kb + 32) * HDIM, 0, 0);
      __builtin_prefetch(Vb + kb + 32, 0, 0);
    }
    // ---- scores: Q[32x64] @ K^T[64x32]; K frags shared by both q-tiles ----
    const v8f z = {0.f,0.f,0.f,0.f,0.f,0.f,0.f,0.f};
    const bf16* Kt = Kb + (size_t)kb * HDIM;
    v16bf kf[2][2];
    kf[0][0] = frag_b_bf16(Kt,                HDIM, lane);
    kf[0][1] = frag_b_bf16(Kt + 32,           HDIM, lane);
    kf[1][0] = frag_b_bf16(Kt + 16*HDIM,      HDIM, lane);
    kf[1][1] = frag_b_bf16(Kt + 16*HDIM + 32, HDIM, lane);
    v8f s[2][2];
#pragma unroll
    for (int q = 0; q < 2; ++q)
#pragma unroll
      for (int j = 0; j < 2; ++j) {
        s[q][j] = wmma_bf16(qa[q][0], kf[j][0], z);
        s[q][j] = wmma_bf16(qa[q][1], kf[j][1], s[q][j]);
      }

    // ---- online softmax (row stats live inside each 16-lane half) --------
#pragma unroll
    for (int q = 0; q < 2; ++q) {
      float corr[8];
#pragma unroll
      for (int v = 0; v < 8; ++v) {
        float rm = fmaxf(s[q][0][v], s[q][1][v]);
#pragma unroll
        for (int m = 1; m < 16; m <<= 1) rm = fmaxf(rm, __shfl_xor(rm, m, 32));
        float mnew = fmaxf(mrun[q][v], rm);
        corr[v] = __expf(mrun[q][v] - mnew);
        mrun[q][v] = mnew;
        float p0 = __expf(s[q][0][v] - mnew);
        float p1 = __expf(s[q][1][v] - mnew);
        s[q][0][v] = p0; s[q][1][v] = p1;
        float rs = p0 + p1;
#pragma unroll
        for (int m = 1; m < 16; m <<= 1) rs += __shfl_xor(rs, m, 32);
        lrun[q][v] = lrun[q][v] * corr[v] + rs;
      }
#pragma unroll
      for (int t = 0; t < 4; ++t)
#pragma unroll
        for (int v = 0; v < 8; ++v) acc[q][t][v] *= corr[v];

      // C-layout -> A-layout via LDS round trip
      bf16* plq = pl + q * 16 * PSTRIDE;
#pragma unroll
      for (int v = 0; v < 8; ++v) {
        int row = v + half * 8;
        plq[row * PSTRIDE + nidx]      = (bf16)s[q][0][v];
        plq[row * PSTRIDE + 16 + nidx] = (bf16)s[q][1][v];
      }
    }
    v16bf pa[2];
    pa[0] = frag_a_bf16(pl,               PSTRIDE, lane);
    pa[1] = frag_a_bf16(pl + 16*PSTRIDE,  PSTRIDE, lane);

    // ---- acc += P[32x32] @ V[32x64]; V frags shared by both q-tiles ------
#pragma unroll
    for (int t = 0; t < 4; ++t) {
      v16bf vb = frag_b_bf16(Vb + (size_t)(t*16)*SEQ + kb, SEQ, lane);
#pragma unroll
      for (int q = 0; q < 2; ++q)
        acc[q][t] = wmma_bf16(pa[q], vb, acc[q][t]);
    }
  }

  // ---- epilogue: normalize, merge heads into Y[N,S,E] ---------------------
#pragma unroll
  for (int q = 0; q < 2; ++q)
#pragma unroll
    for (int v = 0; v < 8; ++v) {
      int qrow = qt * 32 + q * 16 + v + half * 8;
      float invl = 1.0f / lrun[q][v];
      size_t base = ((size_t)n * SEQ + qrow) * E_DIM + h * HDIM;
#pragma unroll
      for (int t = 0; t < 4; ++t)
        Y[base + t*16 + nidx] = (bf16)(acc[q][t][v] * invl);
    }
}

// ---------------------------------------------------------------------------
// Output projection: out = Y(bf16) @ Wo + bo, fp32 result. 64x64 per wave.
// ---------------------------------------------------------------------------
__global__ void __launch_bounds__(256, 2)
out_gemm_kernel(const bf16* __restrict__ Yb, const bf16* __restrict__ Wt,
                const float* __restrict__ bias, float* __restrict__ out,
                int mrows) {
  const int lane = threadIdx.x & 31;
  const int wid  = threadIdx.x >> 5;
  const int gw   = blockIdx.x * 8 + wid;
  const int wm = gw >> 4;
  const int wn = gw & 15;
  if (wm * 64 >= mrows) return;

  const bf16* Yt = Yb + (size_t)wm * 64 * E_DIM;
  const bf16* Wb = Wt + (size_t)wn * 64 * E_DIM;

  v8f acc[4][4];
  gemm_core64<false>(Yt, Wb, lane, acc);

  const int half = lane >> 4, nidx = lane & 15;
#pragma unroll
  for (int j = 0; j < 4; ++j) {
    const int col = wn * 64 + j * 16 + nidx;
    const float bc = bias[col];
#pragma unroll
    for (int i = 0; i < 4; ++i)
#pragma unroll
      for (int v = 0; v < 8; ++v) {
        int row = wm * 64 + i * 16 + v + half * 8;
        out[(size_t)row * E_DIM + col] = acc[i][j][v] + bc;
      }
  }
}

// ---------------------------------------------------------------------------
extern "C" void kernel_launch(void* const* d_in, const int* in_sizes, int n_in,
                              void* d_out, int out_size, void* d_ws, size_t ws_size,
                              hipStream_t stream) {
  (void)in_sizes; (void)n_in; (void)out_size; (void)ws_size;
  const float* query = (const float*)d_in[0];
  const float* key   = (const float*)d_in[1];
  const float* value = (const float*)d_in[2];
  const float* Wq = (const float*)d_in[3];
  const float* bq = (const float*)d_in[4];
  const float* Wk = (const float*)d_in[5];
  const float* bk = (const float*)d_in[6];
  const float* Wv = (const float*)d_in[7];
  const float* bv = (const float*)d_in[8];
  const float* Wo = (const float*)d_in[9];
  const float* bo = (const float*)d_in[10];
  float* out = (float*)d_out;

  const size_t szW = (size_t)E_DIM * E_DIM;          // weight elements
  const size_t szA = (size_t)NBATCH * SEQ * E_DIM;   // activation elements
  char* p = (char*)d_ws;
  bf16* WtQ = (bf16*)p; p += szW * 2;
  bf16* WtK = (bf16*)p; p += szW * 2;
  bf16* WtV = (bf16*)p; p += szW * 2;
  bf16* WtO = (bf16*)p; p += szW * 2;
  bf16* Qh  = (bf16*)p; p += szA * 2;
  bf16* Kh  = (bf16*)p; p += szA * 2;
  bf16* Vt  = (bf16*)p; p += szA * 2;
  bf16* Yb  = (bf16*)p; p += szA * 2;   // 72 MB total workspace

  dim3 blk(256);   // 8 wave32 waves per workgroup

  prep_weight_kernel<<<szW / 256, blk, 0, stream>>>(Wq, WtQ);
  prep_weight_kernel<<<szW / 256, blk, 0, stream>>>(Wk, WtK);
  prep_weight_kernel<<<szW / 256, blk, 0, stream>>>(Wv, WtV);
  prep_weight_kernel<<<szW / 256, blk, 0, stream>>>(Wo, WtO);

  const int mrows = NBATCH * SEQ;                            // 8192
  const int gemmBlocks = (mrows / 64) * (E_DIM / 64) / 8;    // 256
  const float qscale = 0.125f;                               // 1/sqrt(64)
  proj_gemm_kernel<<<gemmBlocks, blk, 0, stream>>>(query, WtQ, bq, Qh, mrows, 0, qscale);
  proj_gemm_kernel<<<gemmBlocks, blk, 0, stream>>>(key,   WtK, bk, Kh, mrows, 1, 1.0f);
  proj_gemm_kernel<<<gemmBlocks, blk, 0, stream>>>(value, WtV, bv, Vt, mrows, 2, 1.0f);

  const int attnBlocks = NBATCH * NHEAD * (SEQ / 32) / 8;    // 512
  flash_attn_kernel<<<attnBlocks, blk, 0, stream>>>(Qh, Kh, Vt, Yb);

  out_gemm_kernel<<<gemmBlocks, blk, 0, stream>>>(Yb, WtO, bo, out, mrows);
}
